// DynamicQuantMatmul_27650999451929
// MI455X (gfx1250) — 8-bit matrix operand order repaired and checked by reading, compiled, not run on hardware
//
#include <hip/hip_runtime.h>
#include <stdint.h>

typedef __attribute__((ext_vector_type(8))) int v8i;

#define M_DIM 8192
#define N_DIM 8192
#define K_DIM 256
#define QMAXF 127.0f
#define EPSF  1e-12f


__global__ void dqmm_init_absmax(unsigned* am) {
    if (threadIdx.x < 2) am[threadIdx.x] = 0u;
}

__global__ void dqmm_absmax(const float* __restrict__ p, int n, unsigned* __restrict__ out) {
    float m = 0.0f;
    int stride = gridDim.x * blockDim.x;
    for (int i = blockIdx.x * blockDim.x + threadIdx.x; i < n; i += stride)
        m = fmaxf(m, fabsf(p[i]));
    #pragma unroll
    for (int off = 16; off > 0; off >>= 1)
        m = fmaxf(m, __shfl_xor(m, off, 32));
    if ((threadIdx.x & 31) == 0)
        atomicMax(out, __float_as_uint(m));
}

__device__ __forceinline__ unsigned quant4(const float v0, const float v1,
                                           const float v2, const float v3, float inv_s) {
    float q0 = fminf(fmaxf(rintf(v0 * inv_s), -QMAXF), QMAXF);
    float q1 = fminf(fmaxf(rintf(v1 * inv_s), -QMAXF), QMAXF);
    float q2 = fminf(fmaxf(rintf(v2 * inv_s), -QMAXF), QMAXF);
    float q3 = fminf(fmaxf(rintf(v3 * inv_s), -QMAXF), QMAXF);
    unsigned d = ((unsigned)((int)q0 & 0xff))
               | ((unsigned)((int)q1 & 0xff) << 8)
               | ((unsigned)((int)q2 & 0xff) << 16)
               | ((unsigned)((int)q3 & 0xff) << 24);
    return d;
}

__global__ void dqmm_quantA(const float* __restrict__ x, const unsigned* __restrict__ am,
                            unsigned* __restrict__ aq) {
    int dw = blockIdx.x * blockDim.x + threadIdx.x;
    float s = fmaxf(__uint_as_float(am[0]) / QMAXF, EPSF);
    float inv_s = 1.0f / s;
    int v      = dw & 7;
    int lane   = (dw >> 3) & 31;
    int blk    = dw >> 8;
    int tile_k = blk & 3;
    int tile_m = blk >> 2;
    int h      = lane >> 4;
    int m      = (tile_m << 4) + (lane & 15);
    int kbase  = (tile_k << 6) + ((v >> 1) << 4) + (h << 3) + ((v & 1) << 2);
    const float* src = x + m * K_DIM + kbase;
    aq[dw] = quant4(src[0], src[1], src[2], src[3], inv_s);
}

__global__ void dqmm_quantB(const float* __restrict__ y, const unsigned* __restrict__ am,
                            unsigned* __restrict__ bq) {
    int dw = blockIdx.x * blockDim.x + threadIdx.x;
    float s = fmaxf(__uint_as_float(am[1]) / QMAXF, EPSF);
    float inv_s = 1.0f / s;
    int v      = dw & 7;
    int lane   = (dw >> 3) & 31;
    int blk    = dw >> 8;
    int tile_k = blk & 3;
    int tile_n = blk >> 2;
    int h      = lane >> 4;
    int n      = (tile_n << 4) + (lane & 15);
    int kbase  = (tile_k << 6) + ((v >> 1) << 4) + (h << 3) + ((v & 1) << 2);
    const float* src = y + kbase * N_DIM + n;
    bq[dw] = quant4(src[0], src[N_DIM], src[2 * N_DIM], src[3 * N_DIM], inv_s);
}

__device__ __forceinline__ void dqmm_stage(const v8i* __restrict__ aq,
                                           const v8i* __restrict__ bq,
                                           v8i* smem, int buf, int k,
                                           int tm_base, int tn_base, int t) {
    const v8i* src;
    if (t < 256) {
        int ti = t >> 5;
        src = aq + (((tm_base + ti) * 4 + k) << 5) + (t & 31);
    } else {
        int tj = (t - 256) >> 5;
        src = bq + (((tn_base + tj) * 4 + k) << 5) + (t & 31);
    }
    unsigned lds = (unsigned)(size_t)&smem[(buf << 9) + t];
    asm volatile(
        "global_load_async_to_lds_b128 %0, %1, off\n\t"
        "global_load_async_to_lds_b128 %0, %1, off offset:16"
        :
        : "v"(lds), "v"(src)
        : "memory");
}

__global__ void __launch_bounds__(512)
dqmm_gemm(const v8i* __restrict__ aq, const v8i* __restrict__ bq,
          const unsigned* __restrict__ am, float* __restrict__ out) {
    __shared__ v8i smem[1024];

    int t    = threadIdx.x;
    int lane = t & 31;
    int wid  = t >> 5;
    int wr   = wid & 3;
    int wc   = wid >> 2;
    int tm_base = blockIdx.y << 3;
    int tn_base = blockIdx.x << 3;
    int tm0  = tm_base + (wr << 1);
    int tn0  = tn_base + (wc << 1);

    float sx = fmaxf(__uint_as_float(am[0]) / QMAXF, EPSF);
    float sy = fmaxf(__uint_as_float(am[1]) / QMAXF, EPSF);
    float scale = sx * sy;

    v8i acc00 = {}, acc01 = {}, acc10 = {}, acc11 = {};

    dqmm_stage(aq, bq, smem, 0, 0, tm_base, tn_base, t);

    int buf = 0;
    #pragma unroll
    for (int k = 0; k < 4; ++k) {
        asm volatile("s_wait_asynccnt 0x0" ::: "memory");
        __syncthreads();
        if (k < 3)
            dqmm_stage(aq, bq, smem, buf ^ 1, k + 1, tm_base, tn_base, t);

        int base = buf << 9;
        v8i a0 = smem[base + ((wr * 2)     << 5) + lane];
        v8i a1 = smem[base + ((wr * 2 + 1) << 5) + lane];
        v8i b0 = smem[base + 256 + ((wc * 2)     << 5) + lane];
        v8i b1 = smem[base + 256 + ((wc * 2 + 1) << 5) + lane];

        acc00 = __builtin_amdgcn_wmma_i32_16x16x64_iu8(true, a0, true, b0, acc00, false, false);
        acc01 = __builtin_amdgcn_wmma_i32_16x16x64_iu8(true, a0, true, b1, acc01, false, false);
        acc10 = __builtin_amdgcn_wmma_i32_16x16x64_iu8(true, a1, true, b0, acc10, false, false);
        acc11 = __builtin_amdgcn_wmma_i32_16x16x64_iu8(true, a1, true, b1, acc11, false, false);
        buf ^= 1;
    }

    int mrow = (lane >> 4) << 3;
    int ncol = lane & 15;
    int r    = (tm0 << 4) + mrow;
    int c    = (tn0 << 4) + ncol;
    #pragma unroll
    for (int j = 0; j < 8; ++j) {
        float* o = out + (long)(r + j) * N_DIM + c;
        o[0]                     = (float)acc00[j] * scale;
        o[16]                    = (float)acc01[j] * scale;
        o[(long)16 * N_DIM]      = (float)acc10[j] * scale;
        o[(long)16 * N_DIM + 16] = (float)acc11[j] * scale;
    }
}

extern "C" void kernel_launch(void* const* d_in, const int* in_sizes, int n_in,
                              void* d_out, int out_size, void* d_ws, size_t ws_size,
                              hipStream_t stream) {
    const float* x = (const float*)d_in[0];
    const float* y = (const float*)d_in[1];
    float* out = (float*)d_out;

    unsigned* am = (unsigned*)d_ws;
    unsigned* aq = (unsigned*)((char*)d_ws + 1024);
    unsigned* bq = (unsigned*)((char*)d_ws + 1024 + (2u << 20));

    dqmm_init_absmax<<<dim3(1), dim3(32), 0, stream>>>(am);
    dqmm_absmax<<<dim3(1024), dim3(256), 0, stream>>>(x, M_DIM * K_DIM, am + 0);
    dqmm_absmax<<<dim3(1024), dim3(256), 0, stream>>>(y, K_DIM * N_DIM, am + 1);
    dqmm_quantA<<<dim3(2048), dim3(256), 0, stream>>>(x, am, aq);
    dqmm_quantB<<<dim3(2048), dim3(256), 0, stream>>>(y, am, bq);
    dqmm_gemm<<<dim3(N_DIM / 128, M_DIM / 128), dim3(512), 0, stream>>>(
        (const v8i*)aq, (const v8i*)bq, am, out);
}
